// RelMultiHeadAttn_55233279427083
// MI455X (gfx1250) — compile-verified, matrix operand order checked by reading, not run on hardware
//
#include <hip/hip_runtime.h>
#include <hip/hip_bf16.h>

// ---------------------------------------------------------------------------
// Transformer-XL RelMultiHeadAttn for MI455X (gfx1250, wave32, WMMA).
//
//   1. f32 -> bf16 conversion of w, Wqkv, r, Wr, Wo            (bandwidth)
//   2. qkv  = w @ Wqkv      [4096x1024 @ 1024x3072] WMMA bf16  (25.8 GF)
//      rk   = r @ Wr        [1024x1024 @ 1024x1024] WMMA bf16  ( 2.1 GF)
//      -> double-buffered GLOBAL_LOAD_ASYNC_TO_LDS pipeline (ASYNCcnt),
//         tile k+1 fetch overlaps tile k WMMA.
//   3. bias_ac[j,n] = sum_{b,d} r_w_bias[n,d]*k[j,b,n,d]   (reference sums b!)
//      d_[j,n]      = sum_d   r_r_bias[n,d]*rk[j,n,d]
//   4. flash attention per (b,n,i-tile): S = QK^T + bias + rel_shift(QR^T + d),
//      online softmax, O += P@V. rel_shift: bd[i,j] = (QR^T)[i, j-i+1023], so
//      each 128x128 tile computes a 128x256 jr window of QR^T via WMMA and
//      scatter-adds along shifted diagonals into the LDS score tile
//      (bijective mapping -> no atomics). ~178 KB LDS fits CDNA5's 320KB/WGP.
//      All tiles staged with async-to-LDS (no VGPR round trip).
//   5. attn_out = attn_vec @ Wo  [4096x1024 @ 1024x1024] WMMA bf16, f32 out
//   6. residual + LayerNorm -> d_out (f32)
// ---------------------------------------------------------------------------

#define QLEN   1024
#define BSZ    4
#define DMODEL 1024
#define NHEAD  16
#define DHEAD  64
#define ND     (NHEAD * DHEAD)   // 1024
#define ROWS   (QLEN * BSZ)      // 4096
#define QKVN   (3 * ND)          // 3072

typedef __attribute__((ext_vector_type(16))) __bf16 bf16x16;
typedef __attribute__((ext_vector_type(8)))  float  f32x8;

__device__ __forceinline__ __bf16 to_bf16(float f) { return (__bf16)f; }
__device__ __forceinline__ float  bf2f(__bf16 h)   { return (float)h; }

__device__ __forceinline__ f32x8 zero8() {
  f32x8 z;
#pragma unroll
  for (int i = 0; i < 8; ++i) z[i] = 0.0f;
  return z;
}

// ---------------------------------------------------------------------------
// CDNA5 async global->LDS copy (ISA 08_async_tensor / VGLOBAL op 98).
// Per-lane: 16 bytes MEM[gsrc] -> LDS[lds_off]. Tracked by ASYNCcnt.
// lds_off is the wave-relative LDS byte address (our kernels' dynamic LDS
// allocations start at offset 0, so plain byte offsets are exact).
// ---------------------------------------------------------------------------
__device__ __forceinline__ void async_ld_b128(unsigned lds_off, const void* gsrc) {
  asm volatile("global_load_async_to_lds_b128 %0, %1, off"
               :: "v"(lds_off), "v"(gsrc)
               : "memory");
}
__device__ __forceinline__ void async_wait0() {
  asm volatile("s_wait_asynccnt 0" ::: "memory");
}

// A fragment: 16x32 bf16, row-major source with row stride `stride` elements.
// ISA 7.12.2: lanes 0-15 row M=lane, K={0..7,16..23}; lanes 16-31 K={8..15,24..31}.
__device__ __forceinline__ bf16x16 load_frag_a(const __bf16* base, int lane, int stride) {
  bf16x16 f;
  const int r  = lane & 15;
  const int hi = (lane >> 4) << 3;  // 0 or 8
  const __bf16* row = base + r * stride;
#pragma unroll
  for (int p = 0; p < 8; ++p) {
    const int kk = 2 * p + ((p >= 4) ? 8 : 0) + hi;
    f[2 * p]     = row[kk];
    f[2 * p + 1] = row[kk + 1];
  }
  return f;
}

// B fragment: 32x16 bf16; element (k,n) at base[n*n_stride + k*k_stride].
__device__ __forceinline__ bf16x16 load_frag_b(const __bf16* base, int lane,
                                               int n_stride, int k_stride) {
  bf16x16 f;
  const int nn = lane & 15;
  const int hi = (lane >> 4) << 3;
  const __bf16* col = base + nn * n_stride;
#pragma unroll
  for (int p = 0; p < 8; ++p) {
    const int kk = 2 * p + ((p >= 4) ? 8 : 0) + hi;
    f[2 * p]     = col[kk * k_stride];
    f[2 * p + 1] = col[(kk + 1) * k_stride];
  }
  return f;
}

__device__ __forceinline__ f32x8 wmma_bf16(bf16x16 a, bf16x16 b, f32x8 c) {
  return __builtin_amdgcn_wmma_f32_16x16x32_bf16(false, a, false, b,
                                                 (short)0, c, false, false);
}

// ---------------------------------------------------------------------------
// f32 -> bf16 conversion
// ---------------------------------------------------------------------------
__global__ __launch_bounds__(256) void rmha_cvt_bf16(const float* __restrict__ in,
                                                     __bf16* __restrict__ out, int n) {
  int i = blockIdx.x * 256 + threadIdx.x;
  if (i < n) out[i] = to_bf16(in[i]);
}

// ---------------------------------------------------------------------------
// Generic bf16 GEMM: C[M,N] = A[M,K] @ B[K,N], row-major, WMMA bf16->f32.
// 128x128x32 block tiles, 256 threads = 8 waves (2x4), each wave a 64x32
// sub-tile (4x2 accumulators, 8 wmma per K-step). Double-buffered async
// global->LDS pipeline. Dynamic LDS: 2x8KB A + 2x8KB B = 32 KB.
// LDS byte map: A[phase] @ phase*8192 ; B[phase] @ 16384 + phase*8192.
// All dims used here are multiples of 128/32 -> no bounds checks.
// ---------------------------------------------------------------------------
__device__ __forceinline__ void gemm_issue_tile(const __bf16* A, const __bf16* B,
                                                int K, int N, int m0, int n0,
                                                int tid, int k0, int phase) {
  const unsigned pa = (unsigned)phase * 8192u;
  const unsigned pb = 16384u + (unsigned)phase * 8192u;
  {  // A tile 128x32: each thread 16 bf16 = 2x b128
    const int r = tid >> 1;
    const int c = (tid & 1) * 16;
    const __bf16* src = A + (size_t)(m0 + r) * K + k0 + c;
    const unsigned dst = pa + (unsigned)(r * 32 + c) * 2u;
    async_ld_b128(dst,       src);
    async_ld_b128(dst + 16u, src + 8);
  }
#pragma unroll
  for (int t = 0; t < 2; ++t) {  // B tile 32x128: each thread 2x b128
    const int cid = tid + t * 256;
    const int r = cid >> 4;
    const int c = (cid & 15) * 8;
    async_ld_b128(pb + (unsigned)(r * 128 + c) * 2u,
                  B + (size_t)(k0 + r) * N + n0 + c);
  }
}

#define GEMM_SMEM 32768

template <bool OUT_BF16>
__global__ __launch_bounds__(256) void rmha_gemm_bf16(const __bf16* __restrict__ A,
                                                      const __bf16* __restrict__ B,
                                                      void* __restrict__ Cout,
                                                      int M, int N, int K) {
  extern __shared__ char gsm[];
  const int tid  = threadIdx.x;
  const int lane = tid & 31;
  const int w    = tid >> 5;
  const int wr   = w >> 2;   // 0..1
  const int wc   = w & 3;    // 0..3
  const int m0   = blockIdx.y * 128;
  const int n0   = blockIdx.x * 128;

  f32x8 acc[4][2];
#pragma unroll
  for (int mi = 0; mi < 4; ++mi)
#pragma unroll
    for (int ni = 0; ni < 2; ++ni) acc[mi][ni] = zero8();

  gemm_issue_tile(A, B, K, N, m0, n0, tid, 0, 0);
  async_wait0();
  __syncthreads();

  const int nk = K / 32;
  for (int kt = 0; kt < nk; ++kt) {
    const int cur = kt & 1;
    if (kt + 1 < nk)  // prefetch next tile into the other buffer (overlaps WMMA)
      gemm_issue_tile(A, B, K, N, m0, n0, tid, (kt + 1) * 32, cur ^ 1);

    const __bf16* sA = (const __bf16*)(gsm + cur * 8192);
    const __bf16* sB = (const __bf16*)(gsm + 16384 + cur * 8192);

    bf16x16 afr[4];
#pragma unroll
    for (int mi = 0; mi < 4; ++mi)
      afr[mi] = load_frag_a(&sA[(wr * 64 + mi * 16) * 32], lane, 32);
#pragma unroll
    for (int ni = 0; ni < 2; ++ni) {
      bf16x16 bfr = load_frag_b(&sB[wc * 32 + ni * 16], lane, /*n_stride=*/1, /*k_stride=*/128);
#pragma unroll
      for (int mi = 0; mi < 4; ++mi)
        acc[mi][ni] = wmma_bf16(afr[mi], bfr, acc[mi][ni]);
    }

    async_wait0();   // my next-tile deposits done
    __syncthreads(); // everyone's reads of `cur` done + deposits visible
  }

  const int lnlo = lane & 15;
  const int hi8  = (lane >> 4) * 8;
#pragma unroll
  for (int mi = 0; mi < 4; ++mi)
#pragma unroll
    for (int ni = 0; ni < 2; ++ni)
#pragma unroll
      for (int p = 0; p < 8; ++p) {
        const int row = m0 + wr * 64 + mi * 16 + p + hi8;
        const int col = n0 + wc * 32 + ni * 16 + lnlo;
        if (OUT_BF16)
          ((__bf16*)Cout)[(size_t)row * N + col] = to_bf16(acc[mi][ni][p]);
        else
          ((float*)Cout)[(size_t)row * N + col] = acc[mi][ni][p];
      }
  (void)M;
}

// ---------------------------------------------------------------------------
// bias_ac[j,n] = sum_{b,d} r_w_bias[n,d] * k[j,b,n,d]   (reference sums batch)
// d_[j,n]      = sum_d     r_r_bias[n,d] * rk[j,n,d]
// ---------------------------------------------------------------------------
__global__ __launch_bounds__(256) void rmha_bias_ac(const __bf16* __restrict__ qkvB,
                                                    const float* __restrict__ rwb,
                                                    float* __restrict__ biasac) {
  const int idx = blockIdx.x * 256 + threadIdx.x;
  if (idx >= QLEN * NHEAD) return;
  const int j = idx >> 4, n = idx & 15;
  float s = 0.0f;
  for (int b = 0; b < BSZ; ++b) {
    const __bf16* kp = qkvB + (size_t)(j * BSZ + b) * QKVN + ND + n * DHEAD;
    for (int d = 0; d < DHEAD; ++d) s += bf2f(kp[d]) * rwb[n * DHEAD + d];
  }
  biasac[idx] = s;
}

__global__ __launch_bounds__(256) void rmha_dvec(const __bf16* __restrict__ rkB,
                                                 const float* __restrict__ rrb,
                                                 float* __restrict__ dvec) {
  const int idx = blockIdx.x * 256 + threadIdx.x;
  if (idx >= QLEN * NHEAD) return;
  const int j = idx >> 4, n = idx & 15;
  const __bf16* rp = rkB + (size_t)j * ND + n * DHEAD;
  float s = 0.0f;
  for (int d = 0; d < DHEAD; ++d) s += bf2f(rp[d]) * rrb[n * DHEAD + d];
  dvec[idx] = s;
}

// ---------------------------------------------------------------------------
// Fused attention. Grid: (8 query tiles, 64 (b,n) heads), 256 threads.
// Dynamic LDS 181760 B (< 320 KB/WGP on CDNA5). LDS byte map:
//   sS 128x128 f32 @ 0        (65536)
//   sM/sL/sC 128 f32 each @ 65536/66048/66560
//   sQ 128x64 bf16 @ 67072    (16384)
//   sK 128x64 bf16 @ 83456
//   sV 128x64 bf16 @ 99840
//   sR 256x64 bf16 @ 116224   (32768)
//   sP 128x128 bf16 @ 148992  (32768)
// ---------------------------------------------------------------------------
#define FO_SQ 67072u
#define FO_SK 83456u
#define FO_SV 99840u
#define FO_SR 116224u
#define FA_SMEM 181760

__global__ __launch_bounds__(256) void rmha_flash_attn(const __bf16* __restrict__ qkvB,
                                                       const __bf16* __restrict__ rkB,
                                                       const float* __restrict__ biasac,
                                                       const float* __restrict__ dvec,
                                                       __bf16* __restrict__ attnV) {
  extern __shared__ char smem[];
  float*  sS = (float*)smem;
  float*  sM = (float*)(smem + 65536);
  float*  sL = (float*)(smem + 66048);
  float*  sC = (float*)(smem + 66560);
  __bf16* sQ = (__bf16*)(smem + FO_SQ);
  __bf16* sK = (__bf16*)(smem + FO_SK);
  __bf16* sV = (__bf16*)(smem + FO_SV);
  __bf16* sR = (__bf16*)(smem + FO_SR);
  __bf16* sP = (__bf16*)(smem + 148992);

  const int tid  = threadIdx.x;
  const int lane = tid & 31;
  const int w    = tid >> 5;
  const int I0   = blockIdx.x * 128;
  const int bn   = blockIdx.y;
  const int b    = bn >> 4;
  const int n    = bn & 15;
  const int lnlo = lane & 15;
  const int hi8  = (lane >> 4) * 8;
  const float scale = 0.125f;  // DHEAD^-0.5

  // Q tile (async): rows i = I0..I0+127, q[i,b,n,:] from qkv part 0
#pragma unroll
  for (int t = 0; t < 4; ++t) {
    const int cid = tid + t * 256, r = cid >> 3, c = (cid & 7) * 8;
    async_ld_b128(FO_SQ + (unsigned)(r * 64 + c) * 2u,
                  qkvB + (size_t)((I0 + r) * BSZ + b) * QKVN + n * DHEAD + c);
  }
  if (tid < 128) { sM[tid] = -3.0e38f; sL[tid] = 0.0f; }

  f32x8 oAcc[4];
#pragma unroll
  for (int ct = 0; ct < 4; ++ct) oAcc[ct] = zero8();

  const int nJ = blockIdx.x + 1;  // causal: j-tiles 0..iTile
  for (int jb = 0; jb < nJ; ++jb) {
    const int J0   = jb * 128;
    const int jrlo = 896 - (I0 - J0);  // low edge of 256-wide jr window, >= 0
    __syncthreads();                   // prior-iter readers done before reload

    // K/V tiles from qkv parts 1/2 (async -> LDS, no VGPR staging)
#pragma unroll
    for (int t = 0; t < 4; ++t) {
      const int cid = tid + t * 256, r = cid >> 3, c = (cid & 7) * 8;
      const size_t rowb = (size_t)((J0 + r) * BSZ + b) * QKVN + n * DHEAD;
      const unsigned off = (unsigned)(r * 64 + c) * 2u;
      async_ld_b128(FO_SK + off, qkvB + rowb + ND + c);
      async_ld_b128(FO_SV + off, qkvB + rowb + 2 * ND + c);
    }
    // rel-pos key window: rows jr = jrlo..jrlo+255 (clamped; overflow is masked)
#pragma unroll
    for (int t = 0; t < 8; ++t) {
      const int cid = tid + t * 256, r = cid >> 3, c = (cid & 7) * 8;
      int jr = jrlo + r; jr = (jr > QLEN - 1) ? (QLEN - 1) : jr;
      async_ld_b128(FO_SR + (unsigned)(r * 64 + c) * 2u,
                    rkB + (size_t)jr * ND + n * DHEAD + c);
    }
    async_wait0();    // my deposits (incl. Q on first iter) complete
    __syncthreads();  // all waves' deposits visible

    // wave w owns score rows w*16..w*16+15
    bf16x16 aq0 = load_frag_a(sQ + (w * 16) * 64, lane, 64);       // d 0..31
    bf16x16 aq1 = load_frag_a(sQ + (w * 16) * 64 + 32, lane, 64);  // d 32..63

    // S_ac = Q K^T (+ bias_ac per column j)
    for (int ct = 0; ct < 8; ++ct) {
      f32x8 acc = zero8();
      bf16x16 b0 = load_frag_b(sK + (ct * 16) * 64, lane, /*n_stride=*/64, /*k_stride=*/1);
      bf16x16 b1 = load_frag_b(sK + (ct * 16) * 64 + 32, lane, 64, 1);
      acc = wmma_bf16(aq0, b0, acc);
      acc = wmma_bf16(aq1, b1, acc);
      const int cg = ct * 16 + lnlo;
      const float bias = biasac[(J0 + cg) * NHEAD + n];
#pragma unroll
      for (int p = 0; p < 8; ++p)
        sS[(w * 16 + p + hi8) * 128 + cg] = acc[p] + bias;
    }
    __syncthreads();

    // S_b = Q R^T over the 256-wide jr window; rel_shift scatter-add:
    //   bd[ir,jc] = Sb[ir, jc - ir + 127]  -> (ir,c) -> (ir, jc=ir+c-127), bijective
    for (int ct = 0; ct < 16; ++ct) {
      f32x8 acc = zero8();
      bf16x16 b0 = load_frag_b(sR + (ct * 16) * 64, lane, 64, 1);
      bf16x16 b1 = load_frag_b(sR + (ct * 16) * 64 + 32, lane, 64, 1);
      acc = wmma_bf16(aq0, b0, acc);
      acc = wmma_bf16(aq1, b1, acc);
      const int c = ct * 16 + lnlo;  // 0..255 within window
      int jr = jrlo + c; jr = (jr > QLEN - 1) ? (QLEN - 1) : jr;
      const float dval = dvec[jr * NHEAD + n];
#pragma unroll
      for (int p = 0; p < 8; ++p) {
        const int ir = w * 16 + p + hi8;
        const int jc = ir + c - 127;
        if (jc >= 0 && jc < 128)
          sS[ir * 128 + jc] += acc[p] + dval;
      }
    }
    __syncthreads();

    // online softmax: thread r < 128 owns score row r
    if (tid < 128) {
      const int r = tid;
      const float mold = sM[r];
      float mx = mold;
      for (int c = 0; c < 128; ++c) {
        float s = sS[r * 128 + c] * scale;
        if (J0 + c > I0 + r) s = -1e30f;  // causal mask (triu k=1)
        mx = fmaxf(mx, s);
      }
      const float corr = __expf(mold - mx);
      float l = sL[r] * corr;
      for (int c = 0; c < 128; ++c) {
        float s = sS[r * 128 + c] * scale;
        if (J0 + c > I0 + r) s = -1e30f;
        const float p = __expf(s - mx);
        l += p;
        sP[r * 128 + c] = to_bf16(p);
      }
      sM[r] = mx; sL[r] = l; sC[r] = corr;
    }
    __syncthreads();

    // rescale O by per-row correction, then O += P @ V
    float crow[8];
#pragma unroll
    for (int p = 0; p < 8; ++p) crow[p] = sC[w * 16 + p + hi8];
#pragma unroll
    for (int ct = 0; ct < 4; ++ct)
#pragma unroll
      for (int p = 0; p < 8; ++p) oAcc[ct][p] *= crow[p];

    for (int kk = 0; kk < 4; ++kk) {
      bf16x16 ap = load_frag_a(sP + (w * 16) * 128 + kk * 32, lane, 128);
#pragma unroll
      for (int ct = 0; ct < 4; ++ct) {
        bf16x16 bv = load_frag_b(sV + (kk * 32) * 64 + ct * 16, lane, /*n_stride=*/1, /*k_stride=*/64);
        oAcc[ct] = wmma_bf16(ap, bv, oAcc[ct]);
      }
    }
  }

  // normalize and write attn_vec (bf16) at [(i*BSZ+b), n*64+d]
#pragma unroll
  for (int ct = 0; ct < 4; ++ct)
#pragma unroll
    for (int p = 0; p < 8; ++p) {
      const int r = w * 16 + p + hi8;
      const float inv = 1.0f / sL[r];
      const int i = I0 + r;
      const int d = ct * 16 + lnlo;
      attnV[(size_t)(i * BSZ + b) * ND + n * DHEAD + d] = to_bf16(oAcc[ct][p] * inv);
    }
}

// ---------------------------------------------------------------------------
// residual + LayerNorm: one block per row (4096 rows x 1024 cols)
// ---------------------------------------------------------------------------
__global__ __launch_bounds__(256) void rmha_residual_ln(const float* __restrict__ wv,
                                                        const float* __restrict__ ao,
                                                        const float* __restrict__ gamma,
                                                        const float* __restrict__ beta,
                                                        float* __restrict__ out) {
  __shared__ float rs[256], rq[256];
  const int row = blockIdx.x;
  const float* wr = wv + (size_t)row * DMODEL;
  const float* ar = ao + (size_t)row * DMODEL;
  float s = 0.0f, q = 0.0f;
  for (int c = threadIdx.x; c < DMODEL; c += 256) {
    const float y = wr[c] + ar[c];
    s += y; q += y * y;
  }
  rs[threadIdx.x] = s; rq[threadIdx.x] = q;
  __syncthreads();
  for (int off = 128; off > 0; off >>= 1) {
    if (threadIdx.x < off) {
      rs[threadIdx.x] += rs[threadIdx.x + off];
      rq[threadIdx.x] += rq[threadIdx.x + off];
    }
    __syncthreads();
  }
  const float mu  = rs[0] * (1.0f / DMODEL);
  const float var = rq[0] * (1.0f / DMODEL) - mu * mu;
  const float inv = rsqrtf(var + 1e-5f);
  for (int c = threadIdx.x; c < DMODEL; c += 256) {
    const float y = wr[c] + ar[c];
    out[(size_t)row * DMODEL + c] = gamma[c] * (y - mu) * inv + beta[c];
  }
}

// ---------------------------------------------------------------------------
// host launch
// ---------------------------------------------------------------------------
extern "C" void kernel_launch(void* const* d_in, const int* in_sizes, int n_in,
                              void* d_out, int out_size, void* d_ws, size_t ws_size,
                              hipStream_t stream) {
  (void)in_sizes; (void)n_in; (void)out_size; (void)ws_size;

  const float* w_in  = (const float*)d_in[0];  // [1024,4,1024]
  const float* r_in  = (const float*)d_in[1];  // [1024,1024]
  // d_in[2] attn_mask: causal triu(k=1), applied analytically
  const float* Wqkv  = (const float*)d_in[3];  // [1024,3072]
  const float* Wr    = (const float*)d_in[4];  // [1024,1024]
  const float* rwb   = (const float*)d_in[5];  // [16,64]
  const float* rrb   = (const float*)d_in[6];  // [16,64]
  const float* Wo    = (const float*)d_in[7];  // [1024,1024]
  const float* gamma = (const float*)d_in[8];
  const float* beta  = (const float*)d_in[9];

  // workspace layout (bytes), total ~70.1 MB
  char* ws = (char*)d_ws;
  __bf16* wB     = (__bf16*)(ws + 0);          // 4096*1024*2 =  8388608
  __bf16* WqkvB  = (__bf16*)(ws + 8388608);    // 1024*3072*2 =  6291456
  __bf16* rB     = (__bf16*)(ws + 14680064);   // 1024*1024*2 =  2097152
  __bf16* WrB    = (__bf16*)(ws + 16777216);   //                2097152
  __bf16* WoB    = (__bf16*)(ws + 18874368);   //                2097152
  __bf16* qkvB   = (__bf16*)(ws + 20971520);   // 4096*3072*2 = 25165824
  __bf16* rkB    = (__bf16*)(ws + 46137344);   //                2097152
  float*  biasac = (float*) (ws + 48234496);   // 1024*16*4   =    65536
  float*  dvec   = (float*) (ws + 48300032);   //                  65536
  __bf16* attnVB = (__bf16*)(ws + 48365568);   // 4096*1024*2 =  8388608
  float*  attnO  = (float*) (ws + 56754176);   // 4096*1024*4 = 16777216

  // 1. convert to bf16
  rmha_cvt_bf16<<<(ROWS * DMODEL + 255) / 256, 256, 0, stream>>>(w_in, wB, ROWS * DMODEL);
  rmha_cvt_bf16<<<(DMODEL * QKVN + 255) / 256, 256, 0, stream>>>(Wqkv, WqkvB, DMODEL * QKVN);
  rmha_cvt_bf16<<<(QLEN * DMODEL + 255) / 256, 256, 0, stream>>>(r_in, rB, QLEN * DMODEL);
  rmha_cvt_bf16<<<(DMODEL * ND + 255) / 256, 256, 0, stream>>>(Wr, WrB, DMODEL * ND);
  rmha_cvt_bf16<<<(ND * DMODEL + 255) / 256, 256, 0, stream>>>(Wo, WoB, ND * DMODEL);

  // 2. projections (WMMA bf16, async double-buffered tiles)
  rmha_gemm_bf16<true><<<dim3(QKVN / 128, ROWS / 128), 256, GEMM_SMEM, stream>>>(
      wB, WqkvB, (void*)qkvB, ROWS, QKVN, DMODEL);
  rmha_gemm_bf16<true><<<dim3(ND / 128, QLEN / 128), 256, GEMM_SMEM, stream>>>(
      rB, WrB, (void*)rkB, QLEN, ND, DMODEL);

  // 3. bias vectors
  rmha_bias_ac<<<(QLEN * NHEAD) / 256, 256, 0, stream>>>(qkvB, rwb, biasac);
  rmha_dvec<<<(QLEN * NHEAD) / 256, 256, 0, stream>>>(rkB, rrb, dvec);

  // 4. fused attention (178 KB dynamic LDS; CDNA5 WGP has 320 KB)
  (void)hipFuncSetAttribute((const void*)rmha_flash_attn,
                            hipFuncAttributeMaxDynamicSharedMemorySize, FA_SMEM);
  rmha_flash_attn<<<dim3(QLEN / 128, BSZ * NHEAD), 256, FA_SMEM, stream>>>(
      qkvB, rkB, biasac, dvec, attnVB);

  // 5. output projection (f32 result)
  rmha_gemm_bf16<false><<<dim3(DMODEL / 128, ROWS / 128), 256, GEMM_SMEM, stream>>>(
      attnVB, WoB, (void*)attnO, ROWS, DMODEL, ND);

  // 6. residual + LayerNorm
  rmha_residual_ln<<<ROWS, 256, 0, stream>>>(w_in, attnO, gamma, beta, (float*)d_out);
}
